// ThermalLSTM_5411658793090
// MI455X (gfx1250) — matrix operand order repaired and checked by reading, not compiled, not run on hardware
//
#include <hip/hip_runtime.h>

// ---------------------------------------------------------------------------
// Fused 2-layer LSTM + FC for MI455X (gfx1250), wave32 + WMMA f16->f32.
// B=131072, T=8, IN=6, H=64, gates G=4H=256.
// ---------------------------------------------------------------------------

typedef _Float16 half16 __attribute__((ext_vector_type(16)));
typedef _Float16 half8  __attribute__((ext_vector_type(8)));
typedef float    floatx8 __attribute__((ext_vector_type(8)));

#define B_TOTAL 131072
#define T_STEPS 8
#define IN_F    6
#define HDIM    64
#define GDIM    256   // 4*H

#define WAVES             4
#define THREADS           (WAVES * 32)
#define SAMPLES_PER_WAVE  16
#define SAMPLES_PER_BLOCK (WAVES * SAMPLES_PER_WAVE)
#define NBLOCKS           (B_TOTAL / SAMPLES_PER_BLOCK)

// ---- fast activations: avoid IEEE divide expansion ------------------------
// gfx1250 has v_tanh_f32; otherwise v_exp_f32 + v_rcp_f32 (single VALU ops).
__device__ __forceinline__ float fast_tanh(float x) {
#if __has_builtin(__builtin_amdgcn_tanhf)
    return __builtin_amdgcn_tanhf(x);
#else
    // exp(-2x) = exp2(-2*log2(e)*x)
    float e = __builtin_amdgcn_exp2f(-2.8853900817779268f * x);
    return (1.0f - e) * __builtin_amdgcn_rcpf(1.0f + e);
#endif
}
__device__ __forceinline__ float fast_sigmoid(float x) {
#if __has_builtin(__builtin_amdgcn_tanhf)
    return __builtin_fmaf(0.5f, __builtin_amdgcn_tanhf(0.5f * x), 0.5f);
#else
    float e = __builtin_amdgcn_exp2f(-1.4426950408889634f * x);
    return __builtin_amdgcn_rcpf(1.0f + e);
#endif
}

__device__ __forceinline__ half16 cat8(half8 a, half8 b) {
    return __builtin_shufflevector(a, b, 0,1,2,3,4,5,6,7,8,9,10,11,12,13,14,15);
}

// WMMA wrapper: D = A(16x32 f16) * B(32x16 f16) + C(16x16 f32)
__device__ __forceinline__ floatx8 wmma_f16(half16 a, half16 b, floatx8 c) {
    return __builtin_amdgcn_wmma_f32_16x16x32_f16(
        /*neg_a=*/false, a, /*neg_b=*/false, b,
        /*c_mod=*/(short)0, c, /*reuse_a=*/false, /*reuse_b=*/false);
}

// A operand (16xK, row-major per-sample rows of pitch P halfs) from LDS.
// Lanes 0-15: M=lane, VGPR halfs 0..15 = K(32kb..+15)
// Lanes 16-31: M=lane-16, halfs 0..15 = K(32kb+16..+31)
template <int P>
__device__ __forceinline__ half16 load_a(const _Float16* A, int kb, int lane) {
    const _Float16* row = A + (lane & 15) * P;
    int lg8 = (lane >> 4) << 4;
    half8 a0 = *(const half8*)(row + kb * 32 + lg8);
    half8 a1 = *(const half8*)(row + kb * 32 + 8 + lg8);
    return cat8(a0, a1);
}

// B operand (KxN=32x16) from LDS weight array stored as rows-per-gate (pitch P
// halfs, row n holds K contiguous). Lanes 0-15: N=n, K=32kb..+15;
// lanes 16-31: N=n, K=32kb+16..+31.
template <int P>
__device__ __forceinline__ half16 load_b(const _Float16* W, int tile, int kb, int lane) {
    const _Float16* row = W + (size_t)(tile * 16 + (lane & 15)) * P;
    int off = kb * 32 + ((lane >> 4) << 4);
    return *(const half16*)(row + off);   // compiler splits into 2x ds_load_b128
}

__global__ void __launch_bounds__(THREADS)
thermal_lstm_fused(const float* __restrict__ x,
                   const float* __restrict__ Wih0, const float* __restrict__ Whh0,
                   const float* __restrict__ bih0, const float* __restrict__ bhh0,
                   const float* __restrict__ Wih1, const float* __restrict__ Whh1,
                   const float* __restrict__ bih1, const float* __restrict__ bhh1,
                   const float* __restrict__ Wfc,  const float* __restrict__ bfc,
                   float* __restrict__ out)
{
    // ---- LDS layout -------------------------------------------------------
    __shared__ __align__(32) _Float16 sWx0[GDIM * 32];    // 16 KB, x-proj padded K=32
    __shared__ __align__(32) _Float16 sWh0[GDIM * 64];    // 32 KB, layer0 recurrent
    __shared__ __align__(32) _Float16 sW1 [GDIM * 128];   // 64 KB, [Wih1 | Whh1]
    __shared__ float sB0[GDIM];                           // bih0+bhh0
    __shared__ float sB1[GDIM];                           // bih1+bhh1
    __shared__ float sWfc[HDIM];
    // per-wave A-operand staging (f16):
    __shared__ __align__(32) _Float16 sAx[WAVES][16 * 32];      // x_t, cols>=6 zero
    __shared__ __align__(32) _Float16 sH0[WAVES][2][16 * 64];   // layer0 h (dbl-buf)
    __shared__ __align__(32) _Float16 sH2[WAVES][2][16 * 64];   // layer1 h (dbl-buf)

    const int tid  = threadIdx.x;
    const int wave = tid >> 5;
    const int lane = tid & 31;
    const int ln   = lane & 15;
    const int lg   = lane >> 4;

    // ---- stage weights (f32 -> f16) ---------------------------------------
    for (int idx = tid; idx < GDIM * 32; idx += THREADS) {
        int r = idx >> 5, k = idx & 31;
        sWx0[idx] = (_Float16)(k < IN_F ? Wih0[r * IN_F + k] : 0.0f);
    }
    for (int idx = tid; idx < GDIM * 64; idx += THREADS) {
        sWh0[idx] = (_Float16)Whh0[idx];
    }
    for (int idx = tid; idx < GDIM * 128; idx += THREADS) {
        int r = idx >> 7, k = idx & 127;
        sW1[idx] = (_Float16)(k < 64 ? Wih1[r * 64 + k] : Whh1[r * 64 + (k - 64)]);
    }
    for (int n = tid; n < GDIM; n += THREADS) {
        sB0[n] = bih0[n] + bhh0[n];
        sB1[n] = bih1[n] + bhh1[n];
    }
    for (int n = tid; n < HDIM; n += THREADS) sWfc[n] = Wfc[n];

    // zero per-wave staging (x pad cols + t=0 recurrent inputs)
    _Float16* ax = sAx[wave];
    for (int i = lane; i < 16 * 32; i += 32) ax[i] = (_Float16)0.0f;
    for (int i = lane; i < 16 * 64; i += 32) {
        sH0[wave][0][i] = (_Float16)0.0f;
        sH2[wave][0][i] = (_Float16)0.0f;
    }
    __syncthreads();

    // ---- per-wave recurrent cell state in registers -----------------------
    float c0[4][8], c1[4][8];
#pragma unroll
    for (int j = 0; j < 4; ++j)
#pragma unroll
        for (int r = 0; r < 8; ++r) { c0[j][r] = 0.0f; c1[j][r] = 0.0f; }

    const long sampleBase = (long)blockIdx.x * SAMPLES_PER_BLOCK + wave * SAMPLES_PER_WAVE;

    for (int t = 0; t < T_STEPS; ++t) {
        const int rd = t & 1, wr = rd ^ 1;

        // stage x_t into padded A rows (lanes 0-15, one sample each)
        if (lane < 16) {
            const float* xr = x + (sampleBase + lane) * (T_STEPS * IN_F) + t * IN_F;
#pragma unroll
            for (int i = 0; i < IN_F; ++i)
                ax[lane * 32 + i] = (_Float16)xr[i];
        }

        const _Float16* h0rd = sH0[wave][rd];
        _Float16*       h0wr = sH0[wave][wr];
        const _Float16* h2rd = sH2[wave][rd];
        _Float16*       h2wr = sH2[wave][wr];

        // ================= layer 0 : g = [x|h0] * [Wx0|Wh0]^T + b ==========
        {
            half16 aX   = load_a<32>(ax,   0, lane);
            half16 aH0a = load_a<64>(h0rd, 0, lane);
            half16 aH0b = load_a<64>(h0rd, 1, lane);
#pragma unroll
            for (int j = 0; j < 4; ++j) {           // hidden chunk: units 16j..16j+15
                floatx8 acc[4];
#pragma unroll
                for (int gsel = 0; gsel < 4; ++gsel) {   // i,f,g,o gate tiles
                    const int tile = gsel * 4 + j;
                    const float bv = sB0[tile * 16 + ln];
                    floatx8 a;
#pragma unroll
                    for (int r = 0; r < 8; ++r) a[r] = bv;
                    a = wmma_f16(aX,   load_b<32>(sWx0, tile, 0, lane), a);
                    a = wmma_f16(aH0a, load_b<64>(sWh0, tile, 0, lane), a);
                    a = wmma_f16(aH0b, load_b<64>(sWh0, tile, 1, lane), a);
                    acc[gsel] = a;
                }
#pragma unroll
                for (int r = 0; r < 8; ++r) {
                    float iv = fast_sigmoid(acc[0][r]);
                    float fv = fast_sigmoid(acc[1][r]);
                    float gv = fast_tanh(acc[2][r]);
                    float ov = fast_sigmoid(acc[3][r]);
                    float c  = fv * c0[j][r] + iv * gv;
                    c0[j][r] = c;
                    float h  = ov * fast_tanh(c);
                    const int m = r + (lg << 3);                 // sample in wave
                    h0wr[m * 64 + j * 16 + ln] = (_Float16)h;    // feeds layer1 + next t
                }
            }
        }

        // ======== layer 1 : g = [h1_t | h2_{t-1}] * [Wih1 | Whh1]^T + b ====
        {
            half16 aH1a = load_a<64>(h0wr, 0, lane);  // h1_t (just written, LDS in-order)
            half16 aH1b = load_a<64>(h0wr, 1, lane);
            half16 aH2a = load_a<64>(h2rd, 0, lane);  // h2_{t-1}
            half16 aH2b = load_a<64>(h2rd, 1, lane);
#pragma unroll
            for (int j = 0; j < 4; ++j) {
                floatx8 acc[4];
#pragma unroll
                for (int gsel = 0; gsel < 4; ++gsel) {
                    const int tile = gsel * 4 + j;
                    const float bv = sB1[tile * 16 + ln];
                    floatx8 a;
#pragma unroll
                    for (int r = 0; r < 8; ++r) a[r] = bv;
                    a = wmma_f16(aH1a, load_b<128>(sW1, tile, 0, lane), a);
                    a = wmma_f16(aH1b, load_b<128>(sW1, tile, 1, lane), a);
                    a = wmma_f16(aH2a, load_b<128>(sW1, tile, 2, lane), a);
                    a = wmma_f16(aH2b, load_b<128>(sW1, tile, 3, lane), a);
                    acc[gsel] = a;
                }
#pragma unroll
                for (int r = 0; r < 8; ++r) {
                    float iv = fast_sigmoid(acc[0][r]);
                    float fv = fast_sigmoid(acc[1][r]);
                    float gv = fast_tanh(acc[2][r]);
                    float ov = fast_sigmoid(acc[3][r]);
                    float c  = fv * c1[j][r] + iv * gv;
                    c1[j][r] = c;
                    float h  = ov * fast_tanh(c);
                    const int m = r + (lg << 3);
                    h2wr[m * 64 + j * 16 + ln] = (_Float16)h;
                }
            }
        }
    }

    // ---- FC head: out[b] = h2_T . Wfc + bfc  (final h2 is in buffer 0) ----
    if (lane < 16) {
        const _Float16* hf = sH2[wave][0] + lane * 64;
        float acc = bfc[0];
#pragma unroll
        for (int n = 0; n < HDIM; ++n)
            acc += (float)hf[n] * sWfc[n];
        out[sampleBase + lane] = acc;
    }
}

extern "C" void kernel_launch(void* const* d_in, const int* in_sizes, int n_in,
                              void* d_out, int out_size, void* d_ws, size_t ws_size,
                              hipStream_t stream) {
    (void)in_sizes; (void)n_in; (void)d_ws; (void)ws_size; (void)out_size;
    const float* x    = (const float*)d_in[0];
    const float* Wih0 = (const float*)d_in[1];
    const float* Whh0 = (const float*)d_in[2];
    const float* bih0 = (const float*)d_in[3];
    const float* bhh0 = (const float*)d_in[4];
    const float* Wih1 = (const float*)d_in[5];
    const float* Whh1 = (const float*)d_in[6];
    const float* bih1 = (const float*)d_in[7];
    const float* bhh1 = (const float*)d_in[8];
    const float* Wfc  = (const float*)d_in[9];
    const float* bfc  = (const float*)d_in[10];
    float* out = (float*)d_out;

    thermal_lstm_fused<<<NBLOCKS, THREADS, 0, stream>>>(
        x, Wih0, Whh0, bih0, bhh0, Wih1, Whh1, bih1, bhh1, Wfc, bfc, out);
}
